// LSTM_84902913507862
// MI455X (gfx1250) — matrix operand order repaired and checked by reading, not compiled, not run on hardware
//
#include <hip/hip_runtime.h>
#include <hip/hip_bf16.h>

typedef __attribute__((ext_vector_type(16))) _Float16 v16h;
typedef __attribute__((ext_vector_type(8)))  _Float16 v8h;
typedef __attribute__((ext_vector_type(4)))  _Float16 v4h;
typedef __attribute__((ext_vector_type(8)))  float    v8f;
typedef __attribute__((ext_vector_type(4)))  float    v4f;

#define HDIM  64
#define GDIM  256      // 4*H gate width
#define TLEN  512
#define BTOT  1024
#define BTILE 16       // batch rows per block == WMMA M
#define HPAD  72       // h LDS row stride (halves), padded vs 64 for bank spread
#define NTHR  128      // 4 waves

// fast device math: single v_exp_f32 / v_rcp_f32, no IEEE divide chains
__device__ __forceinline__ float fast_sigmoid(float x) {
    return __builtin_amdgcn_rcpf(1.0f + __expf(-x));
}
__device__ __forceinline__ float fast_tanh(float x) {
    const float e = __expf(2.0f * x);
    return (e - 1.0f) * __builtin_amdgcn_rcpf(e + 1.0f);
}

// CDNA5 async copy: memory -> LDS, tracked by ASYNCcnt, bypasses VGPRs.
__device__ __forceinline__ void async_b128_to_lds(void* lds_dst, const void* gsrc) {
    const unsigned lds_off = (unsigned)(uintptr_t)lds_dst;  // addr[31:0] == LDS byte address
    asm volatile("global_load_async_to_lds_b128 %0, %1, off"
                 :: "v"(lds_off), "v"(gsrc) : "memory");
}
__device__ __forceinline__ void wait_async0() {
    asm volatile("s_wait_asynccnt 0x0" ::: "memory");
}

// A fragment (16xK, 16-bit): lane<16 -> row=lane, K = kb..kb+15, kb=0; lane>=16 -> kb=16.
__device__ __forceinline__ v16h load_a_frag(const _Float16* base, int lane, int stride, int kt) {
    const int row = lane & 15;
    const int kb  = (lane & 16) ? 16 : 0;
    const _Float16* p = base + row * stride + kt * 32 + kb;
    v8h lo = *(const v8h*)(p);
    v8h hi = *(const v8h*)(p + 8);
    v16h r;
#pragma unroll
    for (int i = 0; i < 8; ++i) { r[i] = lo[i]; r[8 + i] = hi[i]; }
    return r;
}

// B fragment (Kx16): lane<16 -> col N=lane, K=kt*32+0..15 ; lane>=16 -> N=lane-16, K=kt*32+16..31.
// B[k][n] = W[n][k]; W row-major f16 -> 16 contiguous halves per lane (one aligned 32B load).
__device__ __forceinline__ v16h load_b_frag(const _Float16* W, int Ktot, int n0, int lane, int kt) {
    const int n  = lane & 15;
    const int ko = (lane & 16) ? 16 : 0;
    const _Float16* p = W + (size_t)(n0 * 16 + n) * Ktot + kt * 32 + ko;
    return *(const v16h*)(p);
}

// ---------------------------------------------------------------------------
// Fused LSTM layer. 4 waves/block; wave w owns gate N-tiles {w, w+4, w+8, w+12}
// (matching i/f/g/o column slices) so the entire gate pointwise runs in
// registers: i,f,g,o for one LSTM cell sit in the same lane / same VGPR slot.
// One barrier per timestep; input tile double-buffered & pipelined one step.
// ---------------------------------------------------------------------------
template <int IN_K, bool IN_F32, bool OUT_SEQ>
__global__ __launch_bounds__(NTHR)
void lstm_layer_kernel(const void* __restrict__ in_seq,   // f32 [B][T][IN_K] or f16 [B][T][64]
                       const _Float16* __restrict__ Win,  // [256][IN_K] f16 row-major (= Wih)
                       const _Float16* __restrict__ Whh,  // [256][64]  f16 row-major
                       const float* __restrict__ bsum,    // [256] bih+bhh
                       _Float16* __restrict__ hseq_out,   // [B][T][64] f16 (layer0)
                       float* __restrict__ hlast_out)     // [B][64] f32 (layer1)
{
    constexpr int IN_PAD = (IN_K == 32) ? 40 : 72;

    __shared__ __align__(32) _Float16 h_lds[BTILE * HPAD];
    __shared__ __align__(32) _Float16 in_lds[2][BTILE * IN_PAD];

    const int tid   = threadIdx.x;
    const int lane  = tid & 31;
    const int wave  = tid >> 5;            // 0..3
    const int bbase = blockIdx.x * BTILE;

    // ---- weights pinned in VGPRs for all 512 steps ----
    v16h b_in[4][IN_K / 32];
    v16h b_hh[4][2];
    float bias_g[4];
#pragma unroll
    for (int g = 0; g < 4; ++g) {
        const int n0 = wave + g * 4;       // g: 0=i,1=f,2=g,3=o tile for this wave's columns
#pragma unroll
        for (int kt = 0; kt < IN_K / 32; ++kt) b_in[g][kt] = load_b_frag(Win, IN_K, n0, lane, kt);
#pragma unroll
        for (int kt = 0; kt < 2; ++kt)         b_hh[g][kt] = load_b_frag(Whh, HDIM, n0, lane, kt);
        bias_g[g] = bsum[n0 * 16 + (lane & 15)];
    }

    // staging geometry: one 16B transfer per thread per (x8 halves / x4 floats)
    const int srow = tid >> 3;                   // 0..15
    const int scol = (tid & 7) * (IN_F32 ? 4 : 8);
    const float*    xf = (const float*)in_seq;
    const _Float16* xh = (const _Float16*)in_seq;

    for (int i = tid; i < BTILE * HPAD; i += NTHR) h_lds[i] = (_Float16)0.0f;

    // prologue: stage tile for t=0 into buffer 0
    v4f xreg;
    if constexpr (IN_F32) {
        xreg = *(const v4f*)&xf[(size_t)(bbase + srow) * TLEN * IN_K + scol];
        v4h c4;
#pragma unroll
        for (int i = 0; i < 4; ++i) c4[i] = (_Float16)xreg[i];
        *(v4h*)&in_lds[0][srow * IN_PAD + scol] = c4;
    } else {
        async_b128_to_lds(&in_lds[0][srow * IN_PAD + scol],
                          &xh[(size_t)(bbase + srow) * TLEN * IN_K + scol]);
        wait_async0();
    }
    float c_reg[8];
#pragma unroll
    for (int r = 0; r < 8; ++r) c_reg[r] = 0.0f;
    __syncthreads();

    const int rbase = (lane & 16) ? 8 : 0;       // C layout: M = r + rbase
    const int jcol  = wave * 16 + (lane & 15);   // hidden column owned by this lane

    for (int t = 0; t < TLEN; ++t) {
        const int buf = t & 1;

        // ---- issue next step's input transfer NOW (latency hidden behind GEMM) ----
        if constexpr (IN_F32) {
            if (t + 1 < TLEN)
                xreg = *(const v4f*)&xf[(size_t)(bbase + srow) * TLEN * IN_K +
                                        (size_t)(t + 1) * IN_K + scol];
        } else {
            if (t + 1 < TLEN)
                async_b128_to_lds(&in_lds[buf ^ 1][srow * IN_PAD + scol],
                                  &xh[(size_t)(bbase + srow) * TLEN * IN_K +
                                      (size_t)(t + 1) * IN_K + scol]);
        }

        // ---- A fragments (shared across the 4 gate tiles) ----
        v16h a_in[IN_K / 32];
#pragma unroll
        for (int kt = 0; kt < IN_K / 32; ++kt) a_in[kt] = load_a_frag(in_lds[buf], lane, IN_PAD, kt);
        v16h a_h[2];
#pragma unroll
        for (int kt = 0; kt < 2; ++kt) a_h[kt] = load_a_frag(h_lds, lane, HPAD, kt);

        // ---- gates = bias + x_t@Win^T + h@Whh^T : 4 gate tiles via WMMA ----
        v8f acc[4];
#pragma unroll
        for (int g = 0; g < 4; ++g) {
#pragma unroll
            for (int r = 0; r < 8; ++r) acc[g][r] = bias_g[g];
#pragma unroll
            for (int kt = 0; kt < IN_K / 32; ++kt)
                acc[g] = __builtin_amdgcn_wmma_f32_16x16x32_f16(false, a_in[kt], false, b_in[g][kt],
                                                                (short)0, acc[g], false, false);
#pragma unroll
            for (int kt = 0; kt < 2; ++kt)
                acc[g] = __builtin_amdgcn_wmma_f32_16x16x32_f16(false, a_h[kt], false, b_hh[g][kt],
                                                                (short)0, acc[g], false, false);
        }

        // ---- pointwise entirely in registers: 8 LSTM cells per lane ----
#pragma unroll
        for (int r = 0; r < 8; ++r) {
            const float ig = fast_sigmoid(acc[0][r]);
            const float fg = fast_sigmoid(acc[1][r]);
            const float gg = fast_tanh   (acc[2][r]);
            const float og = fast_sigmoid(acc[3][r]);
            const float c  = fg * c_reg[r] + ig * gg;
            c_reg[r] = c;
            const float h  = og * fast_tanh(c);
            const _Float16 h16 = (_Float16)h;
            const int b = r + rbase;
            h_lds[b * HPAD + jcol] = h16;
            if constexpr (OUT_SEQ)
                hseq_out[(size_t)(bbase + b) * TLEN * HDIM + (size_t)t * HDIM + jcol] = h16;
            else if (t == TLEN - 1)
                hlast_out[(size_t)(bbase + b) * HDIM + jcol] = h;
        }

        // ---- land next step's input tile, then the single per-step barrier ----
        if constexpr (IN_F32) {
            if (t + 1 < TLEN) {
                v4h c4;
#pragma unroll
                for (int i = 0; i < 4; ++i) c4[i] = (_Float16)xreg[i];
                *(v4h*)&in_lds[buf ^ 1][srow * IN_PAD + scol] = c4;
            }
        } else {
            wait_async0();   // t+1 asyncs had the whole step to complete
        }
        __syncthreads();
    }
}

// ---------------------------------------------------------------------------
// Weight prep: f32 -> f16 row-major copies + fused bias sums.
// ---------------------------------------------------------------------------
__global__ __launch_bounds__(256)
void prep_kernel(const float* __restrict__ Wih0, const float* __restrict__ Whh0,
                 const float* __restrict__ bih0, const float* __restrict__ bhh0,
                 const float* __restrict__ Wih1, const float* __restrict__ Whh1,
                 const float* __restrict__ bih1, const float* __restrict__ bhh1,
                 _Float16* __restrict__ wih0h, _Float16* __restrict__ whh0h,
                 _Float16* __restrict__ wih1h, _Float16* __restrict__ whh1h,
                 float* __restrict__ bsum0, float* __restrict__ bsum1)
{
    const int tid    = blockIdx.x * blockDim.x + threadIdx.x;
    const int stride = gridDim.x * blockDim.x;
    for (int i = tid; i < GDIM * 32; i += stride) wih0h[i] = (_Float16)Wih0[i];
    for (int i = tid; i < GDIM * 64; i += stride) whh0h[i] = (_Float16)Whh0[i];
    for (int i = tid; i < GDIM * 64; i += stride) wih1h[i] = (_Float16)Wih1[i];
    for (int i = tid; i < GDIM * 64; i += stride) whh1h[i] = (_Float16)Whh1[i];
    for (int i = tid; i < GDIM; i += stride) {
        bsum0[i] = bih0[i] + bhh0[i];
        bsum1[i] = bih1[i] + bhh1[i];
    }
}

// ---------------------------------------------------------------------------
// FC head: out = relu(h_last @ fc1_w^T + fc1_b) @ fc2_w^T + fc2_b   (tiny)
// ---------------------------------------------------------------------------
__global__ __launch_bounds__(128)
void fc_head_kernel(const float* __restrict__ hlast,
                    const float* __restrict__ fc1_w, const float* __restrict__ fc1_b,
                    const float* __restrict__ fc2_w, const float* __restrict__ fc2_b,
                    float* __restrict__ out)
{
    const int b = blockIdx.x * blockDim.x + threadIdx.x;
    if (b >= BTOT) return;
    float h[HDIM];
#pragma unroll
    for (int j = 0; j < HDIM; ++j) h[j] = hlast[(size_t)b * HDIM + j];
    float o0 = fc2_b[0], o1 = fc2_b[1];
#pragma unroll 4
    for (int k = 0; k < 32; ++k) {
        float s = fc1_b[k];
#pragma unroll
        for (int j = 0; j < HDIM; ++j) s = fmaf(h[j], fc1_w[k * HDIM + j], s);
        s = fmaxf(s, 0.0f);
        o0 = fmaf(s, fc2_w[k], o0);
        o1 = fmaf(s, fc2_w[32 + k], o1);
    }
    out[b * 2 + 0] = o0;
    out[b * 2 + 1] = o1;
}

extern "C" void kernel_launch(void* const* d_in, const int* in_sizes, int n_in,
                              void* d_out, int out_size, void* d_ws, size_t ws_size,
                              hipStream_t stream)
{
    (void)in_sizes; (void)n_in; (void)out_size; (void)ws_size;

    const float* x     = (const float*)d_in[0];
    const float* Wih0  = (const float*)d_in[1];
    const float* Whh0  = (const float*)d_in[2];
    const float* bih0  = (const float*)d_in[3];
    const float* bhh0  = (const float*)d_in[4];
    const float* Wih1  = (const float*)d_in[5];
    const float* Whh1  = (const float*)d_in[6];
    const float* bih1  = (const float*)d_in[7];
    const float* bhh1  = (const float*)d_in[8];
    const float* fc1_w = (const float*)d_in[9];
    const float* fc1_b = (const float*)d_in[10];
    const float* fc2_w = (const float*)d_in[11];
    const float* fc2_b = (const float*)d_in[12];

    char* ws = (char*)d_ws;
    size_t off = 0;
    _Float16* h1    = (_Float16*)(ws + off); off += (size_t)BTOT * TLEN * HDIM * sizeof(_Float16); // 64 MB
    float*    hlast = (float*)(ws + off);    off += (size_t)BTOT * HDIM * sizeof(float);           // 256 KB
    _Float16* wih0h = (_Float16*)(ws + off); off += (size_t)GDIM * 32 * sizeof(_Float16);
    _Float16* whh0h = (_Float16*)(ws + off); off += (size_t)GDIM * 64 * sizeof(_Float16);
    _Float16* wih1h = (_Float16*)(ws + off); off += (size_t)GDIM * 64 * sizeof(_Float16);
    _Float16* whh1h = (_Float16*)(ws + off); off += (size_t)GDIM * 64 * sizeof(_Float16);
    float*    bsum0 = (float*)(ws + off);    off += (size_t)GDIM * sizeof(float);
    float*    bsum1 = (float*)(ws + off);    off += (size_t)GDIM * sizeof(float);

    prep_kernel<<<16, 256, 0, stream>>>(Wih0, Whh0, bih0, bhh0, Wih1, Whh1, bih1, bhh1,
                                        wih0h, whh0h, wih1h, whh1h, bsum0, bsum1);

    // Layer 0: x (f32, I=32) -> h1 sequence (f16)
    lstm_layer_kernel<32, true, true><<<BTOT / BTILE, NTHR, 0, stream>>>(
        (const void*)x, wih0h, whh0h, bsum0, h1, nullptr);

    // Layer 1: h1 (f16, I=64) -> h_last (f32), input staged via async-to-LDS
    lstm_layer_kernel<64, false, false><<<BTOT / BTILE, NTHR, 0, stream>>>(
        (const void*)h1, wih1h, whh1h, bsum1, nullptr, hlast);

    fc_head_kernel<<<(BTOT + 127) / 128, 128, 0, stream>>>(
        hlast, fc1_w, fc1_b, fc2_w, fc2_b, (float*)d_out);
}
